// RKA_MultiheadAttention_Fast_22247930593745
// MI455X (gfx1250) — compile-verified, matrix operand order checked by reading, not run on hardware
//
#include <hip/hip_runtime.h>
#include <hip/hip_bf16.h>

// ---------------- problem constants ----------------
#define NB   2
#define LSEQ 2048
#define DMOD 1024
#define HNUM 16
#define DH   64
#define NH   (NB*HNUM)          // 32
#define MROWS (NB*LSEQ)         // 4096
#define QKVN  (3*DMOD)          // 3072

typedef __attribute__((ext_vector_type(8)))  float          v8f;
typedef __attribute__((ext_vector_type(16))) __bf16         v16bf;
typedef __attribute__((ext_vector_type(16))) unsigned short v16u;

// ---------------- helpers ----------------
static __device__ __forceinline__ unsigned short bf16_rne(float f) {
    unsigned int u = __float_as_uint(f);
    u += 0x7FFFu + ((u >> 16) & 1u);   // round-to-nearest-even
    return (unsigned short)(u >> 16);
}
static __device__ __forceinline__ float bf16_to_f(unsigned short h) {
    return __uint_as_float(((unsigned int)h) << 16);
}
static __device__ __forceinline__ void split2(float f, unsigned short& hi, unsigned short& lo) {
    hi = bf16_rne(f);
    lo = bf16_rne(f - bf16_to_f(hi));
}
static __device__ __forceinline__ v8f wmma_bf16(v16bf a, v16bf b, v8f c) {
    return __builtin_amdgcn_wmma_f32_16x16x32_bf16(false, a, false, b, (short)0, c, false, false);
}

// ---------------- split fp32 -> (hi,lo) bf16 ----------------
__global__ void split_kernel(const float* __restrict__ in,
                             unsigned short* __restrict__ hi,
                             unsigned short* __restrict__ lo, int n) {
    int i = blockIdx.x * blockDim.x + threadIdx.x;
    if (i >= n) return;
    unsigned short h, l; split2(in[i], h, l);
    hi[i] = h; lo[i] = l;
}

// ---------------- generic bf16x3 GEMM: C[M,N] = A[M,K] * B[N,K]^T + bias ----------------
// block = 128 threads (4 waves, 2x2), block tile 64x64, wave tile 32x32 (2x2 WMMA tiles)
__global__ __launch_bounds__(128) void gemm_bf16x3(
    int M, int N, int K,
    const unsigned short* __restrict__ Ahi, const unsigned short* __restrict__ Alo,
    const unsigned short* __restrict__ Bhi, const unsigned short* __restrict__ Blo,
    const float* __restrict__ bias, float* __restrict__ C)
{
    const int lane = threadIdx.x & 31;
    const int wave = threadIdx.x >> 5;
    const int wy = wave >> 1, wx = wave & 1;
    const int m0 = blockIdx.y * 64 + wy * 32;
    const int n0 = blockIdx.x * 64 + wx * 32;
    const int r  = lane & 15;
    const int kh = (lane >> 4) * 16;

    v8f acc[2][2] = {};
    for (int k0 = 0; k0 < K; k0 += 32) {
        v16bf ahi[2], alo[2], bhi[2], blo[2];
        #pragma unroll
        for (int i = 0; i < 2; ++i) {
            size_t aoff = (size_t)(m0 + i*16 + r) * K + k0 + kh;
            ahi[i] = *reinterpret_cast<const v16bf*>(Ahi + aoff);
            alo[i] = *reinterpret_cast<const v16bf*>(Alo + aoff);
            size_t boff = (size_t)(n0 + i*16 + r) * K + k0 + kh;
            bhi[i] = *reinterpret_cast<const v16bf*>(Bhi + boff);
            blo[i] = *reinterpret_cast<const v16bf*>(Blo + boff);
        }
        #pragma unroll
        for (int i = 0; i < 2; ++i)
            #pragma unroll
            for (int j = 0; j < 2; ++j) {
                acc[i][j] = wmma_bf16(ahi[i], bhi[j], acc[i][j]);
                acc[i][j] = wmma_bf16(ahi[i], blo[j], acc[i][j]);
                acc[i][j] = wmma_bf16(alo[i], bhi[j], acc[i][j]);
            }
    }
    const int rowoff = (lane >> 4) * 8;
    const int col    = lane & 15;
    #pragma unroll
    for (int i = 0; i < 2; ++i)
        #pragma unroll
        for (int j = 0; j < 2; ++j) {
            int cn = n0 + j*16 + col;
            float bv = bias ? bias[cn] : 0.0f;
            #pragma unroll
            for (int v = 0; v < 8; ++v) {
                int cm = m0 + i*16 + rowoff + v;
                C[(size_t)cm * N + cn] = acc[i][j][v] + bv;
            }
        }
}

// ---------------- conv(kt=3) + exact GELU + LayerNorm(DH) + accumulate q_rel ----------------
// block = (64,4): 64 channels x 4 time steps; grid = NH * L/4
__global__ __launch_bounds__(256) void conv_gelu_ln_kernel(
    const float* __restrict__ qkv, const float* __restrict__ conv_w,
    const float* __restrict__ conv_b, const float* __restrict__ ln_g,
    const float* __restrict__ ln_b, float* __restrict__ qr_sum)
{
    const int c  = threadIdx.x;
    const int tt = threadIdx.y;
    const int head = blockIdx.x >> 9;                 // L/4 = 512 blocks per head
    const int t    = ((blockIdx.x & 511) << 2) + tt;
    const int b  = head >> 4, hh = head & 15;
    const size_t base = (size_t)b * LSEQ * QKVN + hh * DH + c;   // q part: cols [0,D)

    float x = conv_b[c];
    #pragma unroll
    for (int j = 0; j < 3; ++j) {
        int tp = t + j - 2;
        if (tp >= 0) x += qkv[base + (size_t)tp * QKVN] * conv_w[c * 3 + j];
    }
    float g = 0.5f * x * (1.0f + erff(x * 0.70710678118654752f));

    __shared__ float sh[4][64];
    sh[tt][c] = g; __syncthreads();
    #pragma unroll
    for (int off = 32; off > 0; off >>= 1) {
        if (c < off) sh[tt][c] += sh[tt][c + off];
        __syncthreads();
    }
    float mu = sh[tt][0] * (1.0f / 64.0f);
    __syncthreads();
    float d = g - mu;
    sh[tt][c] = d * d; __syncthreads();
    #pragma unroll
    for (int off = 32; off > 0; off >>= 1) {
        if (c < off) sh[tt][c] += sh[tt][c + off];
        __syncthreads();
    }
    float var = sh[tt][0] * (1.0f / 64.0f);
    float ln  = d * rsqrtf(var + 1e-5f) * ln_g[c] + ln_b[c];
    atomicAdd(&qr_sum[head * DH + c], ln);
}

// ---------------- v heads -> transposed bf16 hi/lo: vt[head][d][s] ----------------
__global__ void vtrans_kernel(const float* __restrict__ qkv,
                              unsigned short* __restrict__ vthi,
                              unsigned short* __restrict__ vtlo)
{
    int i = blockIdx.x * blockDim.x + threadIdx.x;      // NH*DH*L = 4194304
    if (i >= NH * DH * LSEQ) return;
    int s = i & (LSEQ - 1);
    int t = i >> 11;
    int d = t & (DH - 1);
    int head = t >> 6;
    int b = head >> 4, hh = head & 15;
    float f = qkv[(size_t)(b * LSEQ + s) * QKVN + 2 * DMOD + hh * DH + d];
    unsigned short h, l; split2(f, h, l);
    vthi[i] = h; vtlo[i] = l;
}

// ---------------- Wk[n,t] = (scale/L) * sum_d qr_sum[n,d]*k[n,t,d]; abs-sum per head ----------------
__global__ __launch_bounds__(256) void wk_kernel(
    const float* __restrict__ qkv, const float* __restrict__ qr_sum,
    float* __restrict__ Wk, float* __restrict__ absum)
{
    const int n = blockIdx.x >> 3;
    const int t = ((blockIdx.x & 7) << 8) + threadIdx.x;
    const int b = n >> 4, hh = n & 15;
    const float* kv = qkv + (size_t)(b * LSEQ + t) * QKVN + DMOD + hh * DH;
    const float* qr = qr_sum + n * DH;
    float dot = 0.0f;
    #pragma unroll 8
    for (int d = 0; d < DH; ++d) dot += qr[d] * kv[d];
    float wk = dot * (0.125f / (float)LSEQ);           // scaling=DH^-0.5, mean over L folded in
    Wk[n * LSEQ + t] = wk;

    __shared__ float sh[256];
    sh[threadIdx.x] = fabsf(wk); __syncthreads();
    #pragma unroll
    for (int off = 128; off > 0; off >>= 1) {
        if (threadIdx.x < off) sh[threadIdx.x] += sh[threadIdx.x + off];
        __syncthreads();
    }
    if (threadIdx.x == 0) atomicAdd(&absum[n], sh[0]);
}

// ---------------- Wn = Wk / (absum + 1e-6), to bf16 hi/lo ----------------
__global__ void wn_kernel(const float* __restrict__ Wk, const float* __restrict__ absum,
                          unsigned short* __restrict__ hi, unsigned short* __restrict__ lo)
{
    int i = blockIdx.x * blockDim.x + threadIdx.x;
    if (i >= NH * LSEQ) return;
    int n = i >> 11;
    float wn = Wk[i] / (absum[n] + 1e-6f);
    unsigned short h, l; split2(wn, h, l);
    hi[i] = h; lo[i] = l;
}

// ---------------- Toeplitz attention GEMM: attn[t,d] = sum_{s<=t} Wn[t-s] v[s,d] ----------------
// per head: M=L, N=DH=64, K=L. grid (1, L/64, NH), block 128. Writes merged hi/lo bf16.
__global__ __launch_bounds__(128) void toeplitz_gemm(
    const unsigned short* __restrict__ Wnhi, const unsigned short* __restrict__ Wnlo,
    const unsigned short* __restrict__ vthi, const unsigned short* __restrict__ vtlo,
    unsigned short* __restrict__ Mhi, unsigned short* __restrict__ Mlo)
{
    const int head = blockIdx.z;
    const int lane = threadIdx.x & 31;
    const int wave = threadIdx.x >> 5;
    const int wy = wave >> 1, wx = wave & 1;
    const int m0 = blockIdx.y * 64 + wy * 32;
    const int n0 = wx * 32;
    const int r  = lane & 15;
    const int kh = (lane >> 4) * 16;
    const unsigned short* wh = Wnhi + head * LSEQ;
    const unsigned short* wl = Wnlo + head * LSEQ;

    v8f acc[2][2] = {};
    const int ksteps = (m0 >> 5) + 1;                  // tiles fully above diagonal are zero
    for (int ks = 0; ks < ksteps; ++ks) {
        const int k0 = ks * 32;
        v16bf ahi[2], alo[2], bhi[2], blo[2];
        #pragma unroll
        for (int i = 0; i < 2; ++i) {
            const int t = m0 + i*16 + r;
            v16u uh, ul;
            #pragma unroll
            for (int j = 0; j < 16; ++j) {
                int idx = t - (k0 + kh + j);
                unsigned short h = 0, l = 0;
                if (idx >= 0) { h = wh[idx]; l = wl[idx]; }
                uh[j] = h; ul[j] = l;
            }
            ahi[i] = __builtin_bit_cast(v16bf, uh);
            alo[i] = __builtin_bit_cast(v16bf, ul);
            size_t boff = ((size_t)head * DH + n0 + i*16 + r) * LSEQ + k0 + kh;
            bhi[i] = *reinterpret_cast<const v16bf*>(vthi + boff);
            blo[i] = *reinterpret_cast<const v16bf*>(vtlo + boff);
        }
        #pragma unroll
        for (int i = 0; i < 2; ++i)
            #pragma unroll
            for (int j = 0; j < 2; ++j) {
                acc[i][j] = wmma_bf16(ahi[i], bhi[j], acc[i][j]);
                acc[i][j] = wmma_bf16(ahi[i], blo[j], acc[i][j]);
                acc[i][j] = wmma_bf16(alo[i], bhi[j], acc[i][j]);
            }
    }
    const int b = head >> 4, hh = head & 15;
    const int rowoff = (lane >> 4) * 8;
    const int col    = lane & 15;
    #pragma unroll
    for (int i = 0; i < 2; ++i)
        #pragma unroll
        for (int j = 0; j < 2; ++j) {
            int d = n0 + j*16 + col;
            #pragma unroll
            for (int v = 0; v < 8; ++v) {
                int t = m0 + i*16 + rowoff + v;
                size_t off = (size_t)(b * LSEQ + t) * DMOD + hh * DH + d;
                unsigned short h, l; split2(acc[i][j][v], h, l);
                Mhi[off] = h; Mlo[off] = l;
            }
        }
}

// ---------------- mock attention weights ----------------
__global__ void mock_kernel(float* __restrict__ out) {
    int i = blockIdx.x * blockDim.x + threadIdx.x;
    if (i < NB * LSEQ) out[i] = 1.0f / (float)LSEQ;
}

// ---------------- host launcher ----------------
extern "C" void kernel_launch(void* const* d_in, const int* in_sizes, int n_in,
                              void* d_out, int out_size, void* d_ws, size_t ws_size,
                              hipStream_t stream) {
    const float* x    = (const float*)d_in[0];   // (2,2048,1024)
    const float* wi   = (const float*)d_in[1];   // (3072,1024)
    const float* bi   = (const float*)d_in[2];   // (3072,)
    const float* cw   = (const float*)d_in[3];   // (64,3)
    const float* cb   = (const float*)d_in[4];   // (64,)
    const float* lng  = (const float*)d_in[5];   // (64,)
    const float* lnb  = (const float*)d_in[6];   // (64,)
    const float* wo   = (const float*)d_in[7];   // (1024,1024)
    const float* bo   = (const float*)d_in[8];   // (1024,)
    float* out = (float*)d_out;

    char* ws = (char*)d_ws;
    size_t off = 0;
    auto take = [&](size_t bytes) { char* p = ws + off; off += (bytes + 255) & ~(size_t)255; return p; };

    float*          qkv   = (float*)         take((size_t)MROWS * QKVN * 4);   // 48 MB
    unsigned short* xhi   = (unsigned short*)take((size_t)MROWS * DMOD * 2);
    unsigned short* xlo   = (unsigned short*)take((size_t)MROWS * DMOD * 2);
    unsigned short* wihi  = (unsigned short*)take((size_t)QKVN * DMOD * 2);
    unsigned short* wilo  = (unsigned short*)take((size_t)QKVN * DMOD * 2);
    unsigned short* owhi  = (unsigned short*)take((size_t)DMOD * DMOD * 2);
    unsigned short* owlo  = (unsigned short*)take((size_t)DMOD * DMOD * 2);
    unsigned short* vthi  = (unsigned short*)take((size_t)NH * DH * LSEQ * 2);
    unsigned short* vtlo  = (unsigned short*)take((size_t)NH * DH * LSEQ * 2);
    unsigned short* mghi  = (unsigned short*)take((size_t)MROWS * DMOD * 2);
    unsigned short* mglo  = (unsigned short*)take((size_t)MROWS * DMOD * 2);
    float*          Wk    = (float*)         take((size_t)NH * LSEQ * 4);
    unsigned short* wnhi  = (unsigned short*)take((size_t)NH * LSEQ * 2);
    unsigned short* wnlo  = (unsigned short*)take((size_t)NH * LSEQ * 2);
    float*          qrsum = (float*)         take((size_t)NH * DH * 4);        // followed by absum
    float*          absum = (float*)         take(256);
    (void)ws_size; (void)in_sizes; (void)n_in; (void)out_size;

    // zero accumulators (qrsum block is padded to 256B, absum directly follows)
    hipMemsetAsync(qrsum, 0, ((size_t)NH * DH * 4 + 255 & ~(size_t)255) + 256, stream);

    // split-precision conversions
    split_kernel<<<(MROWS*DMOD)/256, 256, 0, stream>>>(x,  xhi,  xlo,  MROWS*DMOD);
    split_kernel<<<(QKVN*DMOD)/256,  256, 0, stream>>>(wi, wihi, wilo, QKVN*DMOD);
    split_kernel<<<(DMOD*DMOD)/256,  256, 0, stream>>>(wo, owhi, owlo, DMOD*DMOD);

    // GEMM1: qkv = x @ Win^T + b   (4096 x 3072 x 1024)
    gemm_bf16x3<<<dim3(QKVN/64, MROWS/64), 128, 0, stream>>>(
        MROWS, QKVN, DMOD, xhi, xlo, wihi, wilo, bi, qkv);

    // conv + gelu + layernorm + q_rel accumulation
    conv_gelu_ln_kernel<<<NH * (LSEQ/4), dim3(64,4), 0, stream>>>(qkv, cw, cb, lng, lnb, qrsum);

    // v heads -> transposed bf16 hi/lo
    vtrans_kernel<<<(NH*DH*LSEQ)/256, 256, 0, stream>>>(qkv, vthi, vtlo);

    // Wk and per-head |Wk| sum
    wk_kernel<<<NH * (LSEQ/256), 256, 0, stream>>>(qkv, qrsum, Wk, absum);

    // Wn normalize -> bf16 hi/lo
    wn_kernel<<<(NH*LSEQ)/256, 256, 0, stream>>>(Wk, absum, wnhi, wnlo);

    // Toeplitz causal attention GEMM -> merged (bf16 hi/lo)
    toeplitz_gemm<<<dim3(1, LSEQ/64, NH), 128, 0, stream>>>(wnhi, wnlo, vthi, vtlo, mghi, mglo);

    // GEMM3: out = merged @ Wout^T + b  (4096 x 1024 x 1024) -> d_out
    gemm_bf16x3<<<dim3(DMOD/64, MROWS/64), 128, 0, stream>>>(
        MROWS, DMOD, DMOD, mghi, mglo, owhi, owlo, bo, out);

    // mock attention weights tail
    mock_kernel<<<(NB*LSEQ)/256, 256, 0, stream>>>(out + (size_t)MROWS * DMOD);
}
